// Stage2ModelPT_66451734003971
// MI455X (gfx1250) — matrix operand order repaired and checked by reading, not compiled, not run on hardware
//
#include <hip/hip_runtime.h>

// ---------------- problem constants ----------------
#define BB        64          // batch
#define TT        1000        // timesteps
#define NN        300         // neurons
#define DD        32          // stim dim
#define NP        320         // padded neurons (20 x 16)
#define NWG       10          // persistent workgroups (32 cols each)
#define COLS      32          // output columns per WG
#define NTHREADS  256         // 8 waves of 32
#define DT_C      0.01f
#define UCLIP     35.0f

typedef __attribute__((ext_vector_type(16))) _Float16 v16h;
typedef __attribute__((ext_vector_type(8)))  _Float16 h8;
typedef __attribute__((ext_vector_type(8)))  float    v8f;

__device__ __forceinline__ float sigmoidf_(float x) { return 1.0f / (1.0f + __expf(-x)); }
__device__ __forceinline__ float softplusf_(float x) { return (x > 20.0f) ? x : log1pf(__expf(x)); }

// ---------------- WMMA fragment loaders ----------------
// A-matrix, 16-bit, 16x32 (MxK):
//   lanes 0-15 : M=lane,    elems 0..7 -> K=0..7,  elems 8..15 -> K=16..23
//   lanes16-31 : M=lane-16, elems 0..7 -> K=8..15, elems 8..15 -> K=24..31
__device__ __forceinline__ v16h load_a_h(const _Float16* __restrict__ base,
                                         int row_stride, int m_base, int k_base, int lane) {
  int m   = m_base + (lane & 15);
  int khi = (lane >> 4) ? 8 : 0;
  const _Float16* p = base + (size_t)m * row_stride + k_base + khi;
  h8 c0 = *(const h8*)(p);
  h8 c1 = *(const h8*)(p + 16);
  v16h a;
#pragma unroll
  for (int i = 0; i < 8; ++i) { a[i] = c0[i]; a[i + 8] = c1[i]; }
  return a;
}

// Same A layout but source is f32 (stimulus x), converted in-register.
__device__ __forceinline__ v16h load_a_f32(const float* __restrict__ base,
                                           int row_stride, int m_base, int lane) {
  int m   = m_base + (lane & 15);
  int khi = (lane >> 4) ? 8 : 0;
  const float* p = base + (size_t)m * row_stride + khi;
  v16h a;
#pragma unroll
  for (int i = 0; i < 8; ++i) { a[i] = (_Float16)p[i]; a[i + 8] = (_Float16)p[i + 16]; }
  return a;
}

// B-matrix, 16-bit, 32x16 (KxN), stored as W^T tile [n][k] row-major:
//   lanes 0-15 : N=lane,    K=0..7,16..23;  lanes 16-31: N=lane-16, K=8..15,24..31
__device__ __forceinline__ v16h load_b_h(const _Float16* __restrict__ base,
                                         int row_stride, int n_base, int k_base, int lane) {
  int n = n_base + (lane & 15);
  int k = k_base + ((lane >> 4) ? 8 : 0);
  const _Float16* p = base + (size_t)n * row_stride + k;
  h8 c0 = *(const h8*)(p);
  h8 c1 = *(const h8*)(p + 16);
  v16h b;
#pragma unroll
  for (int i = 0; i < 8; ++i) { b[i] = c0[i]; b[i + 8] = c1[i]; }
  return b;
}

// ---------------- device-wide barrier (persistent grid) ----------------
__device__ __forceinline__ void grid_barrier(unsigned* cnt, unsigned* gen, unsigned nwg) {
  __threadfence();      // release this WG's global writes
  __syncthreads();
  if (threadIdx.x == 0) {
    unsigned g = __hip_atomic_load(gen, __ATOMIC_RELAXED, __HIP_MEMORY_SCOPE_AGENT);
    unsigned a = __hip_atomic_fetch_add(cnt, 1u, __ATOMIC_ACQ_REL, __HIP_MEMORY_SCOPE_AGENT);
    if (a == nwg - 1u) {
      __hip_atomic_store(cnt, 0u, __ATOMIC_RELAXED, __HIP_MEMORY_SCOPE_AGENT);
      __hip_atomic_fetch_add(gen, 1u, __ATOMIC_RELEASE, __HIP_MEMORY_SCOPE_AGENT);
    } else {
      while (__hip_atomic_load(gen, __ATOMIC_ACQUIRE, __HIP_MEMORY_SCOPE_AGENT) == g) {
        __builtin_amdgcn_s_sleep(1);   // 64-clk wake granularity: low barrier latency
      }
    }
  }
  __syncthreads();
  __threadfence();      // acquire: don't serve stale data after the barrier
}

// ---------------- prep kernel 1: f16 weight images (padded, B-layout friendly) ----------------
__global__ void __launch_bounds__(NTHREADS)
prep_w(const float* __restrict__ Wsv_raw, const float* __restrict__ Wdcv_raw,
       const float* __restrict__ Tsv, const float* __restrict__ Tdcv,
       const float* __restrict__ Te, const float* __restrict__ G_raw,
       _Float16* __restrict__ Wsv_h, _Float16* __restrict__ Wdcv_h,
       _Float16* __restrict__ Te_h) {
  int gid = blockIdx.x * blockDim.x + threadIdx.x;
  if (gid >= NP * NP) return;
  int i = gid / NP, k = gid % NP;   // i = post (output row), k = pre (reduction)
  float wsv = 0.f, wdcv = 0.f, te = 0.f;
  if (i < NN && k < NN) {
    int idx = i * NN + k;
    wsv  = softplusf_(Wsv_raw[idx])  * (Tsv[idx]  > 0.f ? 1.f : 0.f);
    wdcv = softplusf_(Wdcv_raw[idx]) * (Tdcv[idx] > 0.f ? 1.f : 0.f);
    float G = sigmoidf_(G_raw[0]) * 2.0f;
    te = G * Te[idx];
  }
  Wsv_h[gid]  = (_Float16)wsv;
  Wdcv_h[gid] = (_Float16)wdcv;
  Te_h[gid]   = (_Float16)te;
}

// ---------------- prep kernel 2: gap-degree, bT image, barrier reset ----------------
__global__ void __launch_bounds__(NTHREADS)
prep_misc(const float* __restrict__ bmat, const float* __restrict__ Te,
          const float* __restrict__ G_raw,
          _Float16* __restrict__ bT_h, float* __restrict__ gdeg,
          unsigned* bar_cnt, unsigned* bar_gen) {
  int gid = blockIdx.x * blockDim.x + threadIdx.x;
  if (gid < NP * DD) {                     // b^T image: [i][d], K=d contiguous
    int i = gid / DD, d = gid % DD;
    bT_h[gid] = (_Float16)((i < NN) ? bmat[i * DD + d] : 0.f);
  }
  if (gid < NP) {                          // G * row-degree of T_e
    float s = 0.f;
    if (gid < NN) for (int k = 0; k < NN; ++k) s += Te[gid * NN + k];
    gdeg[gid] = sigmoidf_(G_raw[0]) * 2.0f * s;
  }
  if (gid == 0) { *bar_cnt = 0u; *bar_gen = 0u; }
}

// ---------------- persistent simulation kernel ----------------
// ONE grid barrier per step: u state is column-private per WG (lives in LDS),
// and the cross-WG activation buffers are ping-pong double-buffered, so only
// the A(t) -> B(t) publication needs device-wide ordering.
__global__ void __launch_bounds__(NTHREADS)
sim_kernel(const float* __restrict__ x,            // [B,T,D]
           const float* __restrict__ u0,           // [B,N]
           const float* __restrict__ lam_raw,      // [N]
           const float* __restrict__ I0,           // [N]
           const int*   __restrict__ sign_t,       // [N]
           const float* __restrict__ a_sv_raw, const float* __restrict__ tau_sv_raw,
           const float* __restrict__ a_dcv_raw, const float* __restrict__ tau_dcv_raw,
           const _Float16* __restrict__ Wsv_h, const _Float16* __restrict__ Wdcv_h,
           const _Float16* __restrict__ Te_h,  const _Float16* __restrict__ bT_h,
           const float* __restrict__ gdeg,
           _Float16* __restrict__ Ysv,             // [2][B][NP] ping-pong
           _Float16* __restrict__ Ydcv,            // [2][B][NP]
           _Float16* __restrict__ Uh,              // [2][B][NP]
           unsigned* bar_cnt, unsigned* bar_gen,
           float* __restrict__ out) {              // [B,T,N]
  __shared__ _Float16 Wsv_l[COLS * NP];            // 20 KB each
  __shared__ _Float16 Wdcv_l[COLS * NP];
  __shared__ _Float16 Te_l[COLS * NP];
  __shared__ _Float16 bT_l[COLS * DD];             // 2 KB
  __shared__ float    s_sv_l[2 * BB * COLS];       // 16 KB (rank-major)
  __shared__ float    s_dcv_l[2 * BB * COLS];      // 16 KB
  __shared__ float    u_l[BB * COLS];              // 8 KB  (column-private u state)
  __shared__ float    sign_l[COLS];

  const int tid    = threadIdx.x;
  const int n_base = blockIdx.x * COLS;            // this WG's output-column base

  // --- load time-invariant weight tiles into LDS; init state ---
  for (int idx = tid; idx < COLS * NP; idx += NTHREADS) {
    int r = idx / NP, k = idx % NP;
    size_t g = (size_t)(n_base + r) * NP + k;
    Wsv_l[idx]  = Wsv_h[g];
    Wdcv_l[idx] = Wdcv_h[g];
    Te_l[idx]   = Te_h[g];
  }
  for (int idx = tid; idx < COLS * DD; idx += NTHREADS) {
    int r = idx / DD, d = idx % DD;
    bT_l[idx] = bT_h[(size_t)(n_base + r) * DD + d];
  }
  for (int idx = tid; idx < 2 * BB * COLS; idx += NTHREADS) {
    s_sv_l[idx] = 0.f; s_dcv_l[idx] = 0.f;
  }
  for (int idx = tid; idx < BB * COLS; idx += NTHREADS) {
    int b = idx >> 5, jl = idx & 31, j = n_base + jl;
    u_l[idx] = (j < NN) ? u0[b * NN + j] : 0.f;
  }
  if (tid < COLS) {
    int j = n_base + tid;
    sign_l[tid] = (j < NN) ? (2.0f * (float)sign_t[j] - 1.0f) : 0.0f;
  }
  __syncthreads();

  // --- per-thread invariant scalars (IIR params) ---
  const float asv0  = sigmoidf_(a_sv_raw[0])  * 10.f;
  const float asv1  = sigmoidf_(a_sv_raw[1])  * 10.f;
  const float adcv0 = sigmoidf_(a_dcv_raw[0]) * 10.f;
  const float adcv1 = sigmoidf_(a_dcv_raw[1]) * 10.f;
  const float alsv0  = __expf(-DT_C / (softplusf_(tau_sv_raw[0])  + 1e-4f));
  const float alsv1  = __expf(-DT_C / (softplusf_(tau_sv_raw[1])  + 1e-4f));
  const float aldcv0 = __expf(-DT_C / (softplusf_(tau_dcv_raw[0]) + 1e-4f));
  const float aldcv1 = __expf(-DT_C / (softplusf_(tau_dcv_raw[1]) + 1e-4f));

  // --- per-lane epilogue invariants (C-matrix layout) ---
  const int wave = tid >> 5, lane = tid & 31;
  const int mt   = (wave & 3) * 16;                // batch-row tile base
  const int ntl  = (wave >> 2) * 16;               // local column tile base (0|16)
  const int cl   = ntl + (lane & 15);              // this lane's local column
  const int i_g  = n_base + cl;                    // this lane's global neuron
  const int rb   = mt + ((lane >> 4) ? 8 : 0);     // batch-row base for acc[r]
  const bool valid = (i_g < NN);
  float lamv = 0.f, i0v = 0.f, gdv = 0.f;
  if (valid) {
    lamv = sigmoidf_(lam_raw[i_g]) * 0.9999f;
    i0v  = I0[i_g];
    gdv  = gdeg[i_g];
  }

  for (int t = 0; t < TT; ++t) {
    const size_t par = (size_t)(t & 1) * (BB * NP);          // ping-pong slot
    _Float16* ysv_w  = Ysv  + par;
    _Float16* ydcv_w = Ydcv + par;
    _Float16* uh_w   = Uh   + par;

    // ---------- Phase A: IIR state update + publish f16 activations ----------
    for (int e = tid; e < BB * COLS; e += NTHREADS) {
      int b = e >> 5, jl = e & 31;
      int j = n_base + jl;
      float u   = u_l[e];
      float phi = sigmoidf_(u);
      float s0 = s_sv_l[e],            s1 = s_sv_l[BB * COLS + e];
      s0 = alsv0 * s0 + (1.f - alsv0) * phi;
      s1 = alsv1 * s1 + (1.f - alsv1) * phi;
      s_sv_l[e] = s0; s_sv_l[BB * COLS + e] = s1;
      float d0 = s_dcv_l[e],           d1 = s_dcv_l[BB * COLS + e];
      d0 = aldcv0 * d0 + (1.f - aldcv0) * phi;
      d1 = aldcv1 * d1 + (1.f - aldcv1) * phi;
      s_dcv_l[e] = d0; s_dcv_l[BB * COLS + e] = d1;
      float sg = sign_l[jl];
      ysv_w[b * NP + j]  = (_Float16)(sg * (asv0 * s0 + asv1 * s1));
      ydcv_w[b * NP + j] = (_Float16)(sg * (adcv0 * d0 + adcv1 * d1));
      uh_w[b * NP + j]   = (_Float16)u;
    }
    grid_barrier(bar_cnt, bar_gen, NWG);             // publish Y(t) to all WGs

    // ---------- Phase B: WMMA GEMMs + epilogue ----------
    v8f acc = {};
#pragma unroll 2
    for (int kt = 0; kt < NP; kt += 32) {
      v16h a0 = load_a_h(ysv_w,  NP, mt, kt, lane);
      v16h b0 = load_b_h(Wsv_l,  NP, ntl, kt, lane);
      acc = __builtin_amdgcn_wmma_f32_16x16x32_f16(false, a0, false, b0, (short)0, acc, false, false);
      v16h a1 = load_a_h(ydcv_w, NP, mt, kt, lane);
      v16h b1 = load_b_h(Wdcv_l, NP, ntl, kt, lane);
      acc = __builtin_amdgcn_wmma_f32_16x16x32_f16(false, a1, false, b1, (short)0, acc, false, false);
      v16h a2 = load_a_h(uh_w,   NP, mt, kt, lane);
      v16h b2 = load_b_h(Te_l,   NP, ntl, kt, lane);
      acc = __builtin_amdgcn_wmma_f32_16x16x32_f16(false, a2, false, b2, (short)0, acc, false, false);
    }
    {   // stimulus: x_t [64,32] @ b^T [32,cols]  (single K tile)
      v16h ax = load_a_f32(x + (size_t)t * DD, TT * DD, mt, lane);
      v16h bx = load_b_h(bT_l, DD, ntl, 0, lane);
      acc = __builtin_amdgcn_wmma_f32_16x16x32_f16(false, ax, false, bx, (short)0, acc, false, false);
    }
#pragma unroll
    for (int r = 0; r < 8; ++r) {
      int b = rb + r;
      float unew = 0.f;
      if (valid) {
        float uo = u_l[b * COLS + cl];
        float I  = acc[r] + i0v - gdv * uo;
        unew = lamv * uo + DT_C * I;
        unew = fminf(UCLIP, fmaxf(-UCLIP, unew));
        out[((size_t)b * TT + t) * NN + i_g] = unew;
      }
      u_l[b * COLS + cl] = unew;                     // padded cols stay 0
    }
    __syncthreads();   // u_l(t) complete before phase A(t+1) reads it (same WG only)
  }
}

// ---------------- host launcher ----------------
extern "C" void kernel_launch(void* const* d_in, const int* in_sizes, int n_in,
                              void* d_out, int out_size, void* d_ws, size_t ws_size,
                              hipStream_t stream) {
  (void)in_sizes; (void)n_in; (void)out_size; (void)ws_size;
  const float* x         = (const float*)d_in[0];
  const float* u0        = (const float*)d_in[1];
  const float* T_e       = (const float*)d_in[2];
  const float* T_sv      = (const float*)d_in[3];
  const float* T_dcv     = (const float*)d_in[4];
  const float* W_sv_raw  = (const float*)d_in[5];
  const float* W_dcv_raw = (const float*)d_in[6];
  const float* lam_raw   = (const float*)d_in[7];
  const float* G_raw     = (const float*)d_in[8];
  const float* I0        = (const float*)d_in[9];
  const float* bmat      = (const float*)d_in[10];
  const float* a_sv_raw  = (const float*)d_in[11];
  const float* tau_sv_raw  = (const float*)d_in[12];
  const float* a_dcv_raw = (const float*)d_in[13];
  const float* tau_dcv_raw = (const float*)d_in[14];
  const int*   sign_t    = (const int*)d_in[15];
  float* out = (float*)d_out;

  // workspace carve-up (256B aligned)
  char* ws = (char*)d_ws;
  size_t off = 0;
  auto take = [&](size_t bytes) -> void* {
    void* p = ws + off;
    off = (off + bytes + 255) & ~(size_t)255;
    return p;
  };
  _Float16* Wsv_h  = (_Float16*)take(NP * NP * sizeof(_Float16));
  _Float16* Wdcv_h = (_Float16*)take(NP * NP * sizeof(_Float16));
  _Float16* Te_h   = (_Float16*)take(NP * NP * sizeof(_Float16));
  _Float16* bT_h   = (_Float16*)take(NP * DD * sizeof(_Float16));
  _Float16* Ysv    = (_Float16*)take(2 * BB * NP * sizeof(_Float16));  // ping-pong
  _Float16* Ydcv   = (_Float16*)take(2 * BB * NP * sizeof(_Float16));
  _Float16* Uh     = (_Float16*)take(2 * BB * NP * sizeof(_Float16));
  float*    gdeg   = (float*)take(NP * sizeof(float));
  unsigned* barc   = (unsigned*)take(2 * sizeof(unsigned));
  unsigned* barg   = barc + 1;

  prep_w<<<(NP * NP + NTHREADS - 1) / NTHREADS, NTHREADS, 0, stream>>>(
      W_sv_raw, W_dcv_raw, T_sv, T_dcv, T_e, G_raw, Wsv_h, Wdcv_h, Te_h);
  prep_misc<<<(NP * DD + NTHREADS - 1) / NTHREADS, NTHREADS, 0, stream>>>(
      bmat, T_e, G_raw, bT_h, gdeg, barc, barg);
  sim_kernel<<<NWG, NTHREADS, 0, stream>>>(
      x, u0, lam_raw, I0, sign_t, a_sv_raw, tau_sv_raw, a_dcv_raw, tau_dcv_raw,
      Wsv_h, Wdcv_h, Te_h, bT_h, gdeg, Ysv, Ydcv, Uh, barc, barg, out);
}
